// Decoder_11673721110692
// MI455X (gfx1250) — matrix operand order repaired and checked by reading, not compiled, not run on hardware
//
#include <hip/hip_runtime.h>
#include <hip/hip_bf16.h>
#include <stddef.h>

// ---------------- problem constants ----------------
#define BB 2048
#define NN 50
#define HH 128
#define DV 8
#define DSF 16
#define DEMB 20
#define GATES 512   // 4*H
#define KCAT 256    // dec_in(128) || h(128)

typedef __attribute__((ext_vector_type(16))) __bf16 bf16x16;
typedef __attribute__((ext_vector_type(8)))  float  floatx8;

// ---------------- math helpers (v_exp_f32 based) ----------------
__device__ __forceinline__ float fast_tanh(float x) {
  float e = __expf(2.0f * x);
  return 1.0f - 2.0f / (e + 1.0f);
}
__device__ __forceinline__ float fast_sigm(float x) {
  return 1.0f / (1.0f + __expf(-x));
}

// ---------------- WMMA fragment loaders (contiguous k order, same in A and B) ----------------
// A matrix 16x32 bf16: lane L -> row m=L&15; lanes 0-15 hold K=0..15,
// lanes 16-31 hold K=16..31; VGPR j holds consecutive pair (2 bf16).
__device__ __forceinline__ bf16x16 load_a_frag(const __bf16* sA, int ldA, int lane, int kbase) {
  int m  = lane & 15;
  int kh = (lane >> 4) * 16;
  const unsigned int* row = (const unsigned int*)(sA + (size_t)m * ldA);
  bf16x16 a;
  unsigned int* ap = (unsigned int*)&a;
#pragma unroll
  for (int j = 0; j < 8; ++j) {
    int k0 = kbase + kh + j * 2;
    ap[j] = row[k0 >> 1];
  }
  return a;
}

// B matrix 32x16 bf16: lane L -> column n=L&15; lanes 0-15 hold K=0..15,
// lanes 16-31 hold K=16..31 (16 consecutive bf16 per lane).
// W is row-major [ncols][ldW] with W[n][k] (i.e. B[k][n] = W[n][k] for X @ W^T).
__device__ __forceinline__ bf16x16 load_b_frag(const __bf16* W, int ldW, int lane, int nbase, int kbase) {
  int n  = nbase + (lane & 15);
  int kh = (lane >> 4) * 16;
  const unsigned int* sp = (const unsigned int*)(W + (size_t)n * ldW + kbase + kh);
  bf16x16 b;
  unsigned int* bp = (unsigned int*)&b;
#pragma unroll
  for (int j = 0; j < 8; ++j) bp[j] = sp[j];
  return b;
}

// D/C 16x16 f32: lane L -> col n=L&15, VGPR r -> row m = r + (L>=16)*8.
// Computes sOut[16][ncols] = sA[16][K] @ W^T + bias0 + bias1  (bf16 in, f32 acc)
__device__ __forceinline__ void gemm_tile16(const __bf16* sA, int K, const __bf16* W,
                                            const float* bias0, const float* bias1,
                                            float* sOut, int ncols, int tid) {
  int wave = tid >> 5, lane = tid & 31;
  int ntiles = ncols >> 4;
  for (int nt = wave; nt < ntiles; nt += 8) {
    int nb = nt * 16;
    floatx8 acc = {};
    for (int kc = 0; kc < K; kc += 32) {
      bf16x16 a = load_a_frag(sA, K, lane, kc);
      bf16x16 b = load_b_frag(W, K, lane, nb, kc);
      acc = __builtin_amdgcn_wmma_f32_16x16x32_bf16(false, a, false, b, (short)0, acc, false, false);
    }
    int col = nb + (lane & 15);
    int mo  = (lane >> 4) * 8;
    float bs = (bias0 ? bias0[col] : 0.0f) + (bias1 ? bias1[col] : 0.0f);
#pragma unroll
    for (int r = 0; r < 8; ++r) sOut[(size_t)(mo + r) * ncols + col] = acc[r] + bs;
  }
}

// Query GEMM [16x128] @ W^T -> [16x128] with B-fragments held in registers.
// Each wave owns exactly one 16-column tile (8 waves x 16 = 128 cols).
__device__ __forceinline__ void gemm_q_reg(const __bf16* sA, const bf16x16* bfrag,
                                           const float* bias, float* sOut, int tid) {
  int wave = tid >> 5, lane = tid & 31;
  int nb = wave * 16;
  floatx8 acc = {};
#pragma unroll
  for (int kc = 0; kc < 4; ++kc) {
    bf16x16 a = load_a_frag(sA, HH, lane, kc * 32);
    acc = __builtin_amdgcn_wmma_f32_16x16x32_bf16(false, a, false, bfrag[kc], (short)0, acc, false, false);
  }
  int col = nb + (lane & 15);
  int mo  = (lane >> 4) * 8;
  float bs = bias[col];
#pragma unroll
  for (int r = 0; r < 8; ++r) sOut[(size_t)(mo + r) * HH + col] = acc[r] + bs;
}

// ---------------- kernel 0: weight f32 -> bf16 conversion ----------------
__global__ __launch_bounds__(256) void wconv_kernel(
    const float* W_ih, const float* W_hh, const float* Wqg, const float* Wqp,
    const float* Wrg, const float* Wrp,
    __bf16* dWcat, __bf16* dWqg, __bf16* dWqp, __bf16* dWrg, __bf16* dWrp) {
  int idx = blockIdx.x * 256 + threadIdx.x;
  if (idx < GATES * KCAT) {                       // Wcat[o][k] = [W_ih | W_hh]
    int o = idx >> 8, k = idx & 255;
    float v = (k < HH) ? W_ih[o * HH + k] : W_hh[o * HH + (k - HH)];
    dWcat[idx] = (__bf16)v;
  } else {
    int i2 = idx - GATES * KCAT;
    int sel = i2 >> 14, rem = i2 & 16383;         // 4 matrices of 128*128
    const float* s = (sel == 0) ? Wqg : (sel == 1) ? Wqp : (sel == 2) ? Wrg : Wrp;
    __bf16* d      = (sel == 0) ? dWqg : (sel == 1) ? dWqp : (sel == 2) ? dWrg : dWrp;
    d[rem] = (__bf16)s[rem];
  }
}

// ---------------- kernel 1: e_g / e_p precompute (WMMA) ----------------
// e[b][n][o] = sum_d Wr[o][d] * cu[n][b][d] + br[o]; cu flat rows are (n*B+b).
__global__ __launch_bounds__(256) void eproj_kernel(
    const float* cu, const __bf16* Wrg, const __bf16* Wrp,
    const float* brg, const float* brp, float* Eg, float* Ep) {
  __shared__ __bf16 s_A[16][HH];
  int tid = threadIdx.x;
  size_t R = (size_t)blockIdx.x * 16;
  for (int idx = tid; idx < 16 * HH; idx += 256) {
    int m = idx >> 7, d = idx & 127;
    s_A[m][d] = (__bf16)cu[(R + m) * HH + d];
  }
  __syncthreads();
  int wave = tid >> 5, lane = tid & 31;
  int nb = wave * 16;
#pragma unroll
  for (int which = 0; which < 2; ++which) {
    const __bf16* W = which ? Wrp : Wrg;
    const float* bias = which ? brp : brg;
    float* E = which ? Ep : Eg;
    floatx8 acc = {};
    for (int kc = 0; kc < HH; kc += 32) {
      bf16x16 a = load_a_frag(&s_A[0][0], HH, lane, kc);
      bf16x16 b = load_b_frag(W, HH, lane, nb, kc);
      acc = __builtin_amdgcn_wmma_f32_16x16x32_bf16(false, a, false, b, (short)0, acc, false, false);
    }
    int o  = nb + (lane & 15);
    int mo = (lane >> 4) * 8;
    float bs = bias[o];
#pragma unroll
    for (int r = 0; r < 8; ++r) {
      size_t row = R + (size_t)(mo + r);
      size_t n_ = row >> 11;          // row / 2048
      size_t b_ = row & 2047;         // row % 2048
      E[(b_ * NN + n_) * HH + o] = acc[r] + bs;
    }
  }
}

// ---------------- attention score helper (wave-parallel over d, shfl reduce) ----
__device__ __forceinline__ void attn_scores(const float (*s_q)[HH], const float* E, const float* v,
                                            const unsigned char (*s_mask)[NN], int r0,
                                            float (*s_u)[NN], int tid, bool tanh10) {
  int wave = tid >> 5, lane = tid & 31;
  for (int p = wave; p < 16 * NN; p += 8) {
    int m = p / NN, n = p % NN;
    bool masked = (s_mask[m][n] != 0);
    float partial = 0.0f;
    if (!masked) {
      const float* e = E + (((size_t)(r0 + m)) * NN + n) * HH;
#pragma unroll
      for (int j = 0; j < 4; ++j) {
        int d = lane + j * 32;
        partial += v[d] * fast_tanh(s_q[m][d] + e[d]);
      }
    }
#pragma unroll
    for (int off = 16; off; off >>= 1) partial += __shfl_xor(partial, off, 32);
    if (lane == 0) {
      float val = tanh10 ? 10.0f * fast_tanh(partial) : partial;
      s_u[m][n] = masked ? -__builtin_inff() : val;
    }
  }
}

// ---------------- kernel 2: persistent decode scan ----------------
__global__ __launch_bounds__(256) void decode_kernel(
    const float* dec0, const float* h0, const float* c0,
    const unsigned char* vmask, const int* start_idx,
    const float* V, const float* start_fea, const float* cu,
    const float* b_ih, const float* b_hh,
    const float* bq_g, const float* v_g,
    const float* bq_p, const float* v_p,
    const float* emb, const float* W_tp, const float* b_tp,
    const __bf16* Wcat, const __bf16* Wqg, const __bf16* Wqp,
    const float* Eg, const float* Ep,
    float* out_logp, float* out_sel, float* out_pred, float* out_pred2) {
  __shared__ float s_h[16][HH];
  __shared__ float s_c[16][HH];
  __shared__ float s_dec[16][HH];
  __shared__ __bf16 s_A[16][KCAT];
  __shared__ float s_gates[16][GATES];
  __shared__ __bf16 s_A2[16][HH];
  __shared__ float s_q[16][HH];
  __shared__ float s_u[16][NN];
  __shared__ float s_a[16][NN];
  __shared__ unsigned char s_mask[16][NN];
  __shared__ int s_sel[16], s_last[16], s_new[16];

  int tid = threadIdx.x;
  int lane = tid & 31;
  int r0 = blockIdx.x * 16;

  // --- hoist loop-invariant query-projection B fragments into registers ---
  // Each wave owns one 16-col tile of Wq_g and Wq_p: 4 K-fragments each.
  bf16x16 fragQg[4], fragQp[4];
  {
    int nb = (tid >> 5) * 16;
#pragma unroll
    for (int kc = 0; kc < 4; ++kc) {
      fragQg[kc] = load_b_frag(Wqg, HH, lane, nb, kc * 32);
      fragQp[kc] = load_b_frag(Wqp, HH, lane, nb, kc * 32);
    }
  }

  // --- init carried state ---
  for (int idx = tid; idx < 16 * HH; idx += 256) {
    int m = idx >> 7, d = idx & 127;
    size_t g = (size_t)(r0 + m) * HH + d;
    s_h[m][d] = h0[g];
    s_c[m][d] = c0[g];
    s_dec[m][d] = dec0[g];
  }
  for (int idx = tid; idx < 16 * NN; idx += 256) {
    int m = idx / NN, n = idx % NN;
    s_mask[m][n] = vmask[(size_t)(r0 + m) * NN + n];
  }
  if (tid < 16) {
    s_sel[tid] = 0;
    s_last[tid] = start_idx[r0 + tid];
  }
  __syncthreads();

  const char* pref_g = (const char*)(Eg + (size_t)r0 * NN * HH);
  const char* pref_p = (const char*)(Ep + (size_t)r0 * NN * HH);
  const size_t E_BYTES = (size_t)16 * NN * HH * sizeof(float);

  for (int t = 0; t < NN; ++t) {
    // --- mask update: scatter (t>0) + mask_modify on last column ---
    if (tid < 16) {
      int row = tid;
      if (t > 0) s_mask[row][s_sel[row]] = 1;
      bool all = true;
      for (int n = 0; n < NN; ++n) all = all && (s_mask[row][n] != 0);
      if (all) s_mask[row][NN - 1] = 0;
    }

    // --- prefetch this step's e_g/e_p streams (global_prefetch_b8) so they
    //     reach WGP$ while the gates GEMM + LSTM elementwise run ---
    for (size_t off = (size_t)tid * 128; off < E_BYTES; off += 256 * 128) {
      __builtin_prefetch(pref_g + off, 0, 1);
      __builtin_prefetch(pref_p + off, 0, 1);
    }
    __syncthreads();

    // --- stage A = [dec_in | h] as bf16 ---
    for (int idx = tid; idx < 16 * KCAT; idx += 256) {
      int m = idx >> 8, k = idx & 255;
      float v = (k < HH) ? s_dec[m][k] : s_h[m][k - HH];
      s_A[m][k] = (__bf16)v;
    }
    __syncthreads();

    // --- LSTM gates GEMM: [16x256] @ Wcat^T -> [16x512] (+ biases) ---
    gemm_tile16(&s_A[0][0], KCAT, Wcat, b_ih, b_hh, &s_gates[0][0], GATES, tid);
    __syncthreads();

    // --- LSTM elementwise ---
    for (int idx = tid; idx < 16 * HH; idx += 256) {
      int m = idx >> 7, d = idx & 127;
      float i_ = fast_sigm(s_gates[m][d]);
      float f_ = fast_sigm(s_gates[m][HH + d]);
      float g_ = fast_tanh(s_gates[m][2 * HH + d]);
      float o_ = fast_sigm(s_gates[m][3 * HH + d]);
      float c2 = f_ * s_c[m][d] + i_ * g_;
      float h2 = o_ * fast_tanh(c2);
      s_c[m][d] = c2;
      s_h[m][d] = h2;
      s_A2[m][d] = (__bf16)h2;
    }
    __syncthreads();

    // --- glimpse query: q = h2 @ Wq_g^T + bq_g (register-resident B) ---
    gemm_q_reg(&s_A2[0][0], fragQg, bq_g, &s_q[0][0], tid);
    __syncthreads();

    // --- glimpse scores + masked softmax ---
    attn_scores(s_q, Eg, v_g, s_mask, r0, s_u, tid, false);
    __syncthreads();
    if (tid < 16) {
      int row = tid;
      float mx = -__builtin_inff();
      for (int n = 0; n < NN; ++n) mx = fmaxf(mx, s_u[row][n]);
      float sum = 0.0f;
      for (int n = 0; n < NN; ++n) {
        float e = __expf(s_u[row][n] - mx);
        s_a[row][n] = e;
        sum += e;
      }
      float inv = 1.0f / sum;
      for (int n = 0; n < NN; ++n) s_a[row][n] *= inv;
    }
    __syncthreads();

    // --- glimpse readout g_l[m][d] = sum_n e_g * a, staged bf16 for pointer q ---
    for (int idx = tid; idx < 16 * HH; idx += 256) {
      int m = idx >> 7, d = idx & 127;
      const float* e = Eg + ((size_t)(r0 + m)) * NN * HH + d;
      float sum = 0.0f;
      for (int n = 0; n < NN; ++n) sum += e[(size_t)n * HH] * s_a[m][n];
      s_A2[m][d] = (__bf16)sum;
    }
    __syncthreads();

    // --- pointer query: q = g_l @ Wq_p^T + bq_p (register-resident B) ---
    gemm_q_reg(&s_A2[0][0], fragQp, bq_p, &s_q[0][0], tid);
    __syncthreads();

    // --- pointer scores: u = 10*tanh(v.tanh(q+e)), masked -inf ---
    attn_scores(s_q, Ep, v_p, s_mask, r0, s_u, tid, true);
    __syncthreads();

    // --- log-softmax, argmax, TP head, output writes (per row) ---
    if (tid < 16) {
      int row = tid;
      size_t b = (size_t)(r0 + row);
      float mx = -__builtin_inff();
      int amax = 0;
      for (int n = 0; n < NN; ++n) {
        float u = s_u[row][n];
        if (u > mx) { mx = u; amax = n; }
      }
      float sum = 0.0f;
      for (int n = 0; n < NN; ++n) sum += __expf(s_u[row][n] - mx);
      float lse = mx + __logf(sum);
      float* lp = out_logp + (b * NN + t) * NN;
      for (int n = 0; n < NN; ++n) lp[n] = s_u[row][n] - lse;

      // time-prediction head
      int li = s_last[row];
      float pred = b_tp[0];
      const float* vl = V + (b * NN + li) * DV;
      const float* vn = V + (b * NN + amax) * DV;
#pragma unroll
      for (int j = 0; j < DV; ++j) pred += W_tp[j] * vl[j];
#pragma unroll
      for (int j = 0; j < DV; ++j) pred += W_tp[DV + j] * vn[j];
#pragma unroll
      for (int j = 0; j < DSF; ++j) pred += W_tp[2 * DV + j] * start_fea[b * DSF + j];
#pragma unroll
      for (int j = 0; j < DEMB; ++j) pred += W_tp[2 * DV + DSF + j] * emb[(size_t)t * DEMB + j];

      out_sel[b * NN + t] = (float)amax;
      out_pred[b * NN + t] = pred;
      out_pred2[b * NN + t] = pred;

      s_new[row] = amax;
      s_sel[row] = amax;
      s_last[row] = amax;
    }
    __syncthreads();

    // --- dec_next = context_update[idx, b] gather ---
    for (int idx = tid; idx < 16 * HH; idx += 256) {
      int m = idx >> 7, d = idx & 127;
      s_dec[m][d] = cu[((size_t)s_new[m] * BB + (r0 + m)) * HH + d];
    }
    __syncthreads();
  }
}

// ---------------- host launcher ----------------
extern "C" void kernel_launch(void* const* d_in, const int* in_sizes, int n_in,
                              void* d_out, int out_size, void* d_ws, size_t ws_size,
                              hipStream_t stream) {
  (void)in_sizes; (void)n_in; (void)out_size; (void)ws_size;
  // inputs (setup_inputs order)
  const float* dec0      = (const float*)d_in[0];
  const float* h0        = (const float*)d_in[1];
  const float* c0        = (const float*)d_in[2];
  const unsigned char* vmask = (const unsigned char*)d_in[3];   // jax bool = 1 byte
  const int*   start_idx = (const int*)d_in[4];
  const float* V         = (const float*)d_in[5];
  const float* start_fea = (const float*)d_in[6];
  const float* cu        = (const float*)d_in[7];
  // d_in[8..12] (b_E, b_E_abs, H_update, V_ft, context) are dead per reference
  const float* W_ih = (const float*)d_in[13];
  const float* W_hh = (const float*)d_in[14];
  const float* b_ih = (const float*)d_in[15];
  const float* b_hh = (const float*)d_in[16];
  const float* Wq_g = (const float*)d_in[17];
  const float* bq_g = (const float*)d_in[18];
  const float* Wr_g = (const float*)d_in[19];
  const float* br_g = (const float*)d_in[20];
  const float* v_g  = (const float*)d_in[21];
  const float* Wq_p = (const float*)d_in[22];
  const float* bq_p = (const float*)d_in[23];
  const float* Wr_p = (const float*)d_in[24];
  const float* br_p = (const float*)d_in[25];
  const float* v_p  = (const float*)d_in[26];
  const float* emb  = (const float*)d_in[27];
  const float* W_tp = (const float*)d_in[28];
  const float* b_tp = (const float*)d_in[29];

  // workspace layout (bytes)
  char* ws = (char*)d_ws;
  const size_t E_ELEMS = (size_t)BB * NN * HH;          // 13,107,200 floats each
  float*  Eg    = (float*)(ws);
  float*  Ep    = (float*)(ws + E_ELEMS * sizeof(float));
  char*   wbase = ws + 2 * E_ELEMS * sizeof(float);
  __bf16* Wcat  = (__bf16*)(wbase);                                   // 512x256
  __bf16* dWqg  = (__bf16*)(wbase + (size_t)GATES * KCAT * 2);        // 128x128
  __bf16* dWqp  = dWqg + (size_t)HH * HH;
  __bf16* dWrg  = dWqp + (size_t)HH * HH;
  __bf16* dWrp  = dWrg + (size_t)HH * HH;

  // outputs: log_p [B,N,N] | sels [B,N] | preds [B,N] | preds[...,None] [B,N,1]
  float* out_logp  = (float*)d_out;
  float* out_sel   = out_logp + (size_t)BB * NN * NN;
  float* out_pred  = out_sel + (size_t)BB * NN;
  float* out_pred2 = out_pred + (size_t)BB * NN;

  // 1) weights -> bf16
  wconv_kernel<<<(GATES * KCAT + 4 * HH * HH) / 256, 256, 0, stream>>>(
      W_ih, W_hh, Wq_g, Wq_p, Wr_g, Wr_p, Wcat, dWqg, dWqp, dWrg, dWrp);
  // 2) e_g / e_p projections (WMMA), L2-resident afterwards (105 MB < 192 MB L2)
  eproj_kernel<<<(NN * BB) / 16, 256, 0, stream>>>(cu, dWrg, dWrp, br_g, br_p, Eg, Ep);
  // 3) persistent 50-step decode, 16 batch rows per workgroup
  decode_kernel<<<BB / 16, 256, 0, stream>>>(
      dec0, h0, c0, vmask, start_idx, V, start_fea, cu,
      b_ih, b_hh, bq_g, v_g, bq_p, v_p, emb, W_tp, b_tp,
      Wcat, dWqg, dWqp, Eg, Ep,
      out_logp, out_sel, out_pred, out_pred2);
}
